// MambaBlock_62448824483844
// MI455X (gfx1250) — compile-verified, matrix operand order checked by reading, not run on hardware
//
#include <hip/hip_runtime.h>
#include <hip/hip_bf16.h>

// ---------------------------------------------------------------------------
// Mamba block for MI455X (gfx1250, wave32, WMMA + TDM, double-buffered LDS).
// f32->bf16 convert passes -> in-GEMM (TDM->LDS ping-pong, bf16 WMMA) ->
// conv+SiLU -> x-proj (wave dot) -> selective scan (writes bf16) -> out-GEMM
// ---------------------------------------------------------------------------

#define D_MODEL 1024
#define D_STATE 16
#define D_CONV  4
#define D_INNER 2048
#define BATCH   2
#define SEQ     1024
#define MROWS   (BATCH * SEQ)      // 2048 GEMM rows
#define XZ_COLS (2 * D_INNER)      // 4096
#define NPROJ   (2 * D_STATE + 1)  // 33

#define BM 64     // block tile M
#define BN 128    // block tile N
#define BK 32     // K step (one WMMA depth)

typedef __attribute__((ext_vector_type(16))) __bf16 v16bf;
typedef __attribute__((ext_vector_type(8)))  __bf16 v8bf;
typedef __attribute__((ext_vector_type(4)))  __bf16 v4bf;
typedef __attribute__((ext_vector_type(8)))  float  v8f;
typedef __attribute__((ext_vector_type(4)))  unsigned int v4u;
typedef __attribute__((ext_vector_type(8)))  int v8i;
typedef __attribute__((ext_vector_type(4)))  int v4i;

static __device__ __forceinline__ __bf16 f2bf(float f) {
    unsigned u = __builtin_bit_cast(unsigned, f);
    u += 0x7FFFu + ((u >> 16) & 1u);          // round-to-nearest-even
    unsigned short h = (unsigned short)(u >> 16);
    return __builtin_bit_cast(__bf16, h);
}

static __device__ __forceinline__ float silu(float v) {
    return v / (1.0f + __expf(-v));
}

static __device__ __forceinline__ unsigned lds_off(const void* p) {
    // generic pointer to LDS: low 32 bits are the LDS byte offset
    return (unsigned)(unsigned long long)(uintptr_t)p;
}

// ---------------------------------------------------------------------------
// TDM: async-copy a 2D bf16 tile (tile_d1 rows x tile_d0 elems) from a
// row-major [tensor_d1, stride0] global tensor into LDS (packed row-major).
// D# layout per CDNA5 ISA; 6-arg builtin
// (uint32x4, int32x8, int32x4, int32x4, int32x8, i32 cpol), trailing
// descriptor groups zero (2D tile: tile_dim2/3/4 = 0 unused).
// ---------------------------------------------------------------------------
static __device__ __forceinline__ void tdm_load_tile_bf16(
        unsigned ldsaddr, const __bf16* gptr,
        unsigned tensor_d0, unsigned tensor_d1,
        unsigned tile_d0, unsigned tile_d1,
        unsigned long long stride0_elems)
{
    unsigned long long ga = (unsigned long long)(uintptr_t)gptr;
    v4u g0;
    g0[0] = 1u;                                   // count=1 (valid user desc)
    g0[1] = ldsaddr;                              // lds_addr [63:32]
    g0[2] = (unsigned)(ga & 0xffffffffu);         // global_addr[31:0]
    g0[3] = (unsigned)((ga >> 32) & 0x01ffffffu)  // global_addr[56:32]
          | (2u << 30);                           // type=2 ("image")
    v8i g1;
    g1[0] = (int)(1u << 16);                      // data_size=1 -> 2 bytes/elem
    g1[1] = (int)((tensor_d0 & 0xffffu) << 16);   // tensor_dim0[15:0] @ bits 63:48
    g1[2] = (int)(((tensor_d0 >> 16) & 0xffffu) |
                  ((tensor_d1 & 0xffffu) << 16)); // dim0 hi | dim1 lo
    g1[3] = (int)(((tensor_d1 >> 16) & 0xffffu) |
                  ((tile_d0 & 0xffffu) << 16));   // dim1 hi | tile_dim0
    g1[4] = (int)(tile_d1 & 0xffffu);             // tile_dim1 (tile_dim2=0)
    g1[5] = (int)(unsigned)(stride0_elems & 0xffffffffu);        // stride0 lo
    g1[6] = (int)(unsigned)((stride0_elems >> 32) & 0xffffu);    // stride0 hi
    g1[7] = 0;                                    // tensor_dim1_stride unused (2D tile)
    v4i gz4 = {0, 0, 0, 0};                       // unused descriptor groups
    v8i gz8 = {0, 0, 0, 0, 0, 0, 0, 0};
    __builtin_amdgcn_tensor_load_to_lds(g0, g1, gz4, gz4, gz8, 0);
}

// ---------------------------------------------------------------------------
// f32 -> bf16 conversion pass (coalesced, 4 elems/thread). n % 4 == 0.
// ---------------------------------------------------------------------------
__global__ __launch_bounds__(256)
void mamba_cvt_bf16(const float* __restrict__ src, __bf16* __restrict__ dst, int n) {
    int i = (blockIdx.x * blockDim.x + threadIdx.x) * 4;
    if (i >= n) return;
    const float4 v = *(const float4*)(src + i);
    v4bf o;
    o[0] = f2bf(v.x); o[1] = f2bf(v.y); o[2] = f2bf(v.z); o[3] = f2bf(v.w);
    *(v4bf*)(dst + i) = o;
}

// ---------------------------------------------------------------------------
// C[M,N] = A[M,K] * B[N,K]^T, bf16 in memory, f32 accumulate.
// Block: 256 threads = 8 waves in 2(M) x 4(N); BM=64, BN=128, BK=32.
// Double-buffered: wave 0 TDM-prefetches K-slab s+1 into the ping-pong LDS
// buffer while all waves run ds_load_b128 fragment reads + 4 WMMAs on slab s;
// the s_wait_tensorcnt at the top of each iteration covers only the previous
// prefetch, so DMA and matrix math overlap.
//
// bf16 fragment layout (ISA 7.12.2): element i of lane L <-> k =
// (i>>3)*16 + (L>>4)*8 + (i&7), row/col = L&15.  C/D f32: VGPR r -> rows
// r / r+8, col = L&15.
// ---------------------------------------------------------------------------
__global__ __launch_bounds__(256)
void mamba_gemm_bf16_wmma(const __bf16* __restrict__ A,
                          const __bf16* __restrict__ Bw,
                          float* __restrict__ C,
                          int M, int N, int K) {
    __shared__ __bf16 ldsA[2][BM * BK];
    __shared__ __bf16 ldsB[2][BN * BK];

    const int wave  = threadIdx.x >> 5;
    const int lane  = threadIdx.x & 31;
    const int waveM = wave >> 2;            // 0..1
    const int waveN = wave & 3;             // 0..3
    const int rowBase = blockIdx.x * BM;    // block M origin
    const int colBase = blockIdx.y * BN;    // block N origin

    const int lrow  = lane & 15;
    const int khalf = (lane >> 4) * 8;      // 0 or 8

    v8f acc[2][2] = {};

    const unsigned ldsA_off[2] = { lds_off(ldsA[0]), lds_off(ldsA[1]) };
    const unsigned ldsB_off[2] = { lds_off(ldsB[0]), lds_off(ldsB[1]) };

    const int steps = K / BK;
    if (wave == 0) {                        // prologue: fill buffer 0
        tdm_load_tile_bf16(ldsA_off[0], A + (size_t)rowBase * K,
                           (unsigned)K, (unsigned)M, BK, BM,
                           (unsigned long long)K);
        tdm_load_tile_bf16(ldsB_off[0], Bw + (size_t)colBase * K,
                           (unsigned)K, (unsigned)N, BK, BN,
                           (unsigned long long)K);
    }

    for (int s = 0; s < steps; ++s) {
        const int buf = s & 1;
        if (wave == 0)
            __builtin_amdgcn_s_wait_tensorcnt(0);   // buffer `buf` DMA done
        __syncthreads();                            // visible to all waves;
                                                    // buf^1 free to overwrite
        if (wave == 0 && s + 1 < steps) {           // prefetch next K-slab
            const int kn = (s + 1) * BK;
            tdm_load_tile_bf16(ldsA_off[buf ^ 1], A + (size_t)rowBase * K + kn,
                               (unsigned)K, (unsigned)M, BK, BM,
                               (unsigned long long)K);
            tdm_load_tile_bf16(ldsB_off[buf ^ 1], Bw + (size_t)colBase * K + kn,
                               (unsigned)K, (unsigned)N, BK, BN,
                               (unsigned long long)K);
        }

        v16bf af[2], bf[2];
#pragma unroll
        for (int t = 0; t < 2; ++t) {
            const __bf16* ar = &ldsA[buf][(waveM * 32 + t * 16 + lrow) * BK + khalf];
            v8bf a0 = *(const v8bf*)ar;
            v8bf a1 = *(const v8bf*)(ar + 16);
            const __bf16* br = &ldsB[buf][(waveN * 32 + t * 16 + lrow) * BK + khalf];
            v8bf b0 = *(const v8bf*)br;
            v8bf b1 = *(const v8bf*)(br + 16);
#pragma unroll
            for (int i = 0; i < 8; ++i) {
                af[t][i] = a0[i]; af[t][8 + i] = a1[i];
                bf[t][i] = b0[i]; bf[t][8 + i] = b1[i];
            }
        }
#pragma unroll
        for (int tm = 0; tm < 2; ++tm)
#pragma unroll
            for (int tn = 0; tn < 2; ++tn)
                acc[tm][tn] = __builtin_amdgcn_wmma_f32_16x16x32_bf16(
                    false, af[tm], false, bf[tn],
                    (short)0, acc[tm][tn], false, false);
    }

    const int outcol0 = colBase + waveN * 32 + (lane & 15);
    const int outrow0 = rowBase + waveM * 32 + (lane >> 4) * 8;
#pragma unroll
    for (int tm = 0; tm < 2; ++tm)
#pragma unroll
        for (int tn = 0; tn < 2; ++tn) {
            const int orow = outrow0 + tm * 16;
            const int ocol = outcol0 + tn * 16;
#pragma unroll
            for (int r = 0; r < 8; ++r)
                C[(size_t)(orow + r) * N + ocol] = acc[tm][tn][r];
        }
}

// ---------------------------------------------------------------------------
// Causal depthwise conv (K=4) + bias + SiLU on the x_ssm half of xz.
// ---------------------------------------------------------------------------
__global__ __launch_bounds__(256)
void mamba_conv_silu(const float* __restrict__ xz,
                     const float* __restrict__ conv_w,
                     const float* __restrict__ conv_b,
                     float* __restrict__ xc) {
    int idx = blockIdx.x * blockDim.x + threadIdx.x;   // over B*L*D_INNER
    if (idx >= BATCH * SEQ * D_INNER) return;
    int d = idx % D_INNER;
    int l = (idx / D_INNER) % SEQ;
    int b = idx / (D_INNER * SEQ);

    float acc = conv_b[d];
#pragma unroll
    for (int k = 0; k < D_CONV; ++k) {
        int lt = l - (D_CONV - 1) + k;
        if (lt >= 0)
            acc += xz[((size_t)(b * SEQ + lt)) * XZ_COLS + d] * conv_w[d * D_CONV + k];
    }
    xc[idx] = silu(acc);
}

// ---------------------------------------------------------------------------
// proj[m, j] = dot(xc[m, :], W_x[j, :])  (N=33 -> wave-per-output reduction)
// ---------------------------------------------------------------------------
__global__ __launch_bounds__(32)
void mamba_xproj(const float* __restrict__ xc,
                 const float* __restrict__ Wx,
                 float* __restrict__ proj) {
    int m = blockIdx.x;
    int j = blockIdx.y;
    int lane = threadIdx.x;
    const float* xr = xc + (size_t)m * D_INNER;
    const float* wr = Wx + (size_t)j * D_INNER;
    float acc = 0.f;
    for (int k = lane; k < D_INNER; k += 32)
        acc += xr[k] * wr[k];
#pragma unroll
    for (int off = 16; off > 0; off >>= 1)
        acc += __shfl_down(acc, off, 32);
    if (lane == 0)
        proj[(size_t)m * NPROJ + j] = acc;
}

// ---------------------------------------------------------------------------
// Selective scan: one thread per (b,d) channel, h[16] in VGPRs, sequential
// over L. Fuses dt/softplus, dA, dBx, state update, C-contraction, D skip,
// SiLU(z) gate, and the f32->bf16 store for the out-GEMM.
// ---------------------------------------------------------------------------
__global__ __launch_bounds__(256)
void mamba_scan(const float* __restrict__ xz,     // for z half
                const float* __restrict__ xc,
                const float* __restrict__ proj,
                const float* __restrict__ dt_w,
                const float* __restrict__ dt_b,
                const float* __restrict__ A_log,
                const float* __restrict__ D_param,
                __bf16* __restrict__ ybf) {
    int d = blockIdx.x * blockDim.x + threadIdx.x;
    int b = blockIdx.y;
    if (d >= D_INNER) return;

    float Acoef[D_STATE];
#pragma unroll
    for (int n = 0; n < D_STATE; ++n)
        Acoef[n] = -__expf(A_log[d * D_STATE + n]);

    float h[D_STATE];
#pragma unroll
    for (int n = 0; n < D_STATE; ++n) h[n] = 0.f;

    const float dw = dt_w[d];
    const float db = dt_b[d];
    const float Dp = D_param[d];

    for (int t = 0; t < SEQ; ++t) {
        const size_t m = (size_t)b * SEQ + t;
        const float* pr = proj + m * NPROJ;
        const float xcv = xc[m * D_INNER + d];
        const float zv  = xz[m * XZ_COLS + D_INNER + d];

        const float dtr = pr[0] * dw + db;
        const float dt  = (dtr > 20.f) ? dtr : __logf(1.f + __expf(dtr)); // softplus

        float yv = 0.f;
#pragma unroll
        for (int n = 0; n < D_STATE; ++n) {
            const float dA  = __expf(dt * Acoef[n]);
            const float dBx = dt * pr[1 + n] * xcv;
            h[n] = dA * h[n] + dBx;
            yv  += h[n] * pr[1 + D_STATE + n];
        }
        yv += xcv * Dp;
        ybf[m * D_INNER + d] = f2bf(yv * silu(zv));
    }
}

// ---------------------------------------------------------------------------
// Launch
// ---------------------------------------------------------------------------
extern "C" void kernel_launch(void* const* d_in, const int* in_sizes, int n_in,
                              void* d_out, int out_size, void* d_ws, size_t ws_size,
                              hipStream_t stream) {
    const float* x       = (const float*)d_in[0];
    const float* W_in    = (const float*)d_in[1];
    const float* conv_w  = (const float*)d_in[2];
    const float* conv_b  = (const float*)d_in[3];
    const float* W_x     = (const float*)d_in[4];
    const float* dt_w    = (const float*)d_in[5];
    const float* dt_b    = (const float*)d_in[6];
    const float* A_log   = (const float*)d_in[7];
    const float* D_param = (const float*)d_in[8];
    const float* W_out   = (const float*)d_in[9];
    float* out = (float*)d_out;

    // Workspace carve-up: f32 xz | xc | proj, then bf16 x | W_in | W_out | y
    float* ws    = (float*)d_ws;
    float* xz    = ws;                                  // 2048*4096 f32
    float* xc    = xz + (size_t)MROWS * XZ_COLS;        // 2048*2048 f32
    float* proj  = xc + (size_t)MROWS * D_INNER;        // 2048*33 f32
    __bf16* xbf   = (__bf16*)(proj + (size_t)MROWS * NPROJ);
    __bf16* Winbf = xbf   + (size_t)MROWS * D_MODEL;    // 2048*1024
    __bf16* Woutbf= Winbf + (size_t)XZ_COLS * D_MODEL;  // 4096*1024
    __bf16* ybf   = Woutbf+ (size_t)D_MODEL * D_INNER;  // 1024*2048

    // 0) one-time f32 -> bf16 conversion passes (bandwidth-bound, coalesced)
    {
        int nx = MROWS * D_MODEL;
        mamba_cvt_bf16<<<(nx / 4 + 255) / 256, 256, 0, stream>>>(x, xbf, nx);
        int nw = XZ_COLS * D_MODEL;
        mamba_cvt_bf16<<<(nw / 4 + 255) / 256, 256, 0, stream>>>(W_in, Winbf, nw);
        int no = D_MODEL * D_INNER;
        mamba_cvt_bf16<<<(no / 4 + 255) / 256, 256, 0, stream>>>(W_out, Woutbf, no);
    }
    // 1) xz = x @ W_in^T   (M=2048, N=4096, K=1024)
    {
        dim3 grid(MROWS / BM, XZ_COLS / BN);
        mamba_gemm_bf16_wmma<<<grid, 256, 0, stream>>>(xbf, Winbf, xz,
                                                       MROWS, XZ_COLS, D_MODEL);
    }
    // 2) xc = SiLU(causal_dwconv(x_ssm) + b)
    {
        int total = BATCH * SEQ * D_INNER;
        mamba_conv_silu<<<(total + 255) / 256, 256, 0, stream>>>(xz, conv_w, conv_b, xc);
    }
    // 3) proj = xc @ W_x^T  (N=33, wave-per-dot)
    {
        dim3 grid(MROWS, NPROJ);
        mamba_xproj<<<grid, 32, 0, stream>>>(xc, W_x, proj);
    }
    // 4) selective scan + D skip + SiLU(z) gate -> y (bf16)
    {
        dim3 grid(D_INNER / 256, BATCH);
        mamba_scan<<<grid, 256, 0, stream>>>(xz, xc, proj, dt_w, dt_b,
                                             A_log, D_param, ybf);
    }
    // 5) out = y @ W_out^T  (M=2048, N=1024, K=2048)
    {
        dim3 grid(MROWS / BM, D_MODEL / BN);
        mamba_gemm_bf16_wmma<<<grid, 256, 0, stream>>>(ybf, Woutbf, out,
                                                       MROWS, D_MODEL, D_INNER);
    }
}
